// SelfAttention_G_13451837571122
// MI455X (gfx1250) — matrix operand order repaired and checked by reading, not compiled, not run on hardware
//
#include <hip/hip_runtime.h>
#include <cstdint>
#include <cstddef>

// ---------------------------------------------------------------------------
// CDNA5 (gfx1250, wave32) body/hand GCN stack.
// All dense matmuls -> v_wmma_f32_16x16x32_f16 (f32 accumulate).
// Weights pre-packed into WMMA fragment order (f16) once per launch; A tiles
// staged in LDS as f16 so the inner loop is 2x ds_load_b128 + 2x global_b128
// + 1x wmma.
// ---------------------------------------------------------------------------

typedef __attribute__((ext_vector_type(16))) _Float16 v16h;
typedef __attribute__((ext_vector_type(8)))  _Float16 v8h;
typedef __attribute__((ext_vector_type(8)))  float    v8f;

#define G_TOT 4096     // B*T = 64*64 graphs
#define JD    64
#define HEADS 4
#define NB    10
#define NH    42

__constant__ int c_par_body[NB] = {-1,0,1,2,3,2,5,6,2,8};
__constant__ int c_par_hand[NH] = {
  -1,0,1,2,3,0,5,6,7,0,9,10,11,0,13,14,15,0,17,18,19,
  -1,21,22,23,24,21,26,27,28,21,30,31,32,21,34,35,36,21,38,39,40};

__device__ __forceinline__ float leaky02(float x) { return x > 0.f ? x : 0.2f * x; }

// ---------------------------------------------------------------------------
// x = transpose(refined_feats, (0,2,1)) : [64,256,64] -> [4096, 256]
// ---------------------------------------------------------------------------
__global__ void transpose_btc_kernel(const float* __restrict__ src,
                                     float* __restrict__ X, int total)
{
  int idx = blockIdx.x * blockDim.x + threadIdx.x;
  if (idx >= total) return;
  int c = idx & 255;
  int g = idx >> 8;        // g = b*64 + t
  int b = g >> 6;
  int t = g & 63;
  X[idx] = src[(size_t)b * 16384 + (size_t)c * 64 + t];
}

// ---------------------------------------------------------------------------
// Pack W[K,N] (f32 row-major) into WMMA B-fragment order (f16):
//   frag(nt,kt): lane = kg*16 + n16 holds 16 contiguous halves, K=kt*32+kg*16+j
//   Bp[(((nt*KT + kt)*32 + lane)*16 + j]
// ---------------------------------------------------------------------------
__global__ void pack_w_kernel(const float* __restrict__ W, _Float16* __restrict__ Bp,
                              int K, int N, int total)
{
  int idx = blockIdx.x * blockDim.x + threadIdx.x;   // idx = k*N + n
  if (idx >= total) return;
  int n = idx % N;
  int k = idx / N;
  int nt  = n >> 4, n16 = n & 15;
  int kt  = k >> 5;
  int kg  = (k >> 4) & 1;
  int j   = k & 15;
  int lane = kg * 16 + n16;
  size_t o = (((size_t)nt * (K >> 5) + kt) * 32 + lane) * 16 + j;
  Bp[o] = (_Float16)W[idx];
}

// ---------------------------------------------------------------------------
// WMMA GEMM:  C[M,N] = A[M,K] @ W[K,N] (+bias) (+=C if ACC)
// A: f32 row-major (converted to f16 in LDS), Bp: fragment-packed f16.
// Shapes guaranteed: M%16==0, N%64==0, K%32==0.
// Block = 128 threads = 4 waves; block tile 16(M) x 64(N); wave tile 16x16.
// ---------------------------------------------------------------------------
template<int ACC>
__global__ __launch_bounds__(128)
void gemm_wmma_kernel(const float* __restrict__ A, const _Float16* __restrict__ Bp,
                      const float* __restrict__ bias, float* __restrict__ C,
                      int M, int K, int N)
{
  __shared__ _Float16 sA[16 * 32];     // 16 rows x 32 K (f16), 1 KiB

  const int tid   = threadIdx.x;
  const int wave  = tid >> 5;
  const int lane  = tid & 31;
  const int m0    = blockIdx.x * 16;
  const int ntile = blockIdx.y * 4 + wave;
  const int n0    = ntile * 16;
  const int kg    = lane >> 4;         // K half-group
  const int l16   = lane & 15;
  const int KT    = K >> 5;

  // this wave's packed B fragments: advance 32 v16h per kt
  const v16h* bfrag = (const v16h*)Bp + (size_t)ntile * KT * 32 + lane;

  // this thread's slice of the cooperative A-tile load
  const int ar  = tid >> 3;            // 0..15
  const int akc = (tid & 7) * 4;       // 0..28
  const float* aptr = A + (size_t)(m0 + ar) * K + akc;

  v8f acc = {};

  for (int kt = 0; kt < KT; ++kt) {
    // cooperative A tile load + f32->f16 convert into LDS
    {
      float4 v = *(const float4*)(aptr + kt * 32);
      _Float16* d = sA + ar * 32 + akc;
      d[0] = (_Float16)v.x; d[1] = (_Float16)v.y;
      d[2] = (_Float16)v.z; d[3] = (_Float16)v.w;
    }
    if (kt + 1 < KT)
      __builtin_prefetch(aptr + (kt + 1) * 32, 0, 1);   // global_prefetch_b8
    __syncthreads();

    // A fragment: two aligned 16B LDS reads per lane
    v16h af;
    {
      v8h lo = *(const v8h*)(sA + l16 * 32 + kg * 16);
      v8h hi = *(const v8h*)(sA + l16 * 32 + kg * 16 + 8);
#pragma unroll
      for (int j = 0; j < 8; ++j) { af[j] = lo[j]; af[j + 8] = hi[j]; }
    }
    // B fragment: one contiguous 32B coalesced global read per lane
    v16h bf = bfrag[(size_t)kt * 32];

    acc = __builtin_amdgcn_wmma_f32_16x16x32_f16(
        false, af, false, bf, (short)0, acc, false, false);

    __syncthreads();
  }

  // store: C/D layout: VGPR r, lanes 0-15 -> M=r, lanes 16-31 -> M=8+r
  const float bv = bias ? bias[n0 + l16] : 0.0f;
#pragma unroll
  for (int r = 0; r < 8; ++r) {
    size_t idx = (size_t)(m0 + kg * 8 + r) * N + n0 + l16;
    float v = acc[r] + bv;
    if (ACC) v += C[idx];
    C[idx] = v;
  }
}

// ---------------------------------------------------------------------------
// GAT attention (per graph): Hh = [CH*N, 256] chunk-local ([n][h][f]),
// scores s = h . a, masked softmax over neighbors, head-mean, + bias.
// ---------------------------------------------------------------------------
__global__ __launch_bounds__(256)
void gat_attn_kernel(const float* __restrict__ Hh,
                     const float* __restrict__ asrc,  // [4,64]
                     const float* __restrict__ adst,  // [4,64]
                     const float* __restrict__ bias,  // [64]
                     float* __restrict__ Y,
                     int Nj, int g_off, int is_hand)
{
  __shared__ float s_src[NH * HEADS];
  __shared__ float s_dst[NH * HEADS];
  __shared__ float s_att[NH * HEADS * NH];   // [(i*4+h)*Nj + j]

  const int gl  = blockIdx.x;
  const int tid = threadIdx.x;
  const int* par = is_hand ? c_par_hand : c_par_body;
  const float* hg = Hh + (size_t)gl * Nj * 256;

  for (int idx = tid; idx < Nj * HEADS; idx += blockDim.x) {
    int n = idx >> 2, h = idx & 3;
    const float* hp = hg + n * 256 + h * 64;
    const float* as = asrc + h * 64;
    const float* ad = adst + h * 64;
    float ss = 0.f, sd = 0.f;
    for (int f = 0; f < 64; ++f) { float v = hp[f]; ss += v * as[f]; sd += v * ad[f]; }
    s_src[idx] = ss; s_dst[idx] = sd;
  }
  __syncthreads();

  for (int idx = tid; idx < Nj * HEADS; idx += blockDim.x) {
    int i = idx >> 2, h = idx & 3;
    int pi = par[i];
    float sd = s_dst[idx];
    float mx = -3.4e38f;
    for (int j = 0; j < Nj; ++j) {
      bool m = (i == j) || (pi == j) || (par[j] == i);
      if (m) mx = fmaxf(mx, leaky02(sd + s_src[j * 4 + h]));
    }
    float* arow = s_att + (size_t)idx * Nj;
    float sum = 0.f;
    for (int j = 0; j < Nj; ++j) {
      bool m = (i == j) || (pi == j) || (par[j] == i);
      float w = m ? __expf(leaky02(sd + s_src[j * 4 + h]) - mx) : 0.f;
      arow[j] = w; sum += w;
    }
    float inv = 1.f / sum;   // self-loop guarantees sum > 0
    for (int j = 0; j < Nj; ++j) arow[j] *= inv;
  }
  __syncthreads();

  for (int idx = tid; idx < Nj * 64; idx += blockDim.x) {
    int i = idx >> 6, f = idx & 63;
    float accv = 0.f;
    for (int h = 0; h < HEADS; ++h) {
      const float* arow = s_att + (size_t)(i * 4 + h) * Nj;
      const float* hcol = hg + h * 64 + f;
      for (int j = 0; j < Nj; ++j) accv += arow[j] * hcol[(size_t)j * 256];
    }
    Y[((size_t)(g_off + gl) * Nj + i) * 64 + f] = accv * 0.25f + bias[f];
  }
}

// ---------------------------------------------------------------------------
// GraphConv aggregation: AH[g,i,f] = sum_{j ~ i} H[g,j,f]  (no self loops)
// ---------------------------------------------------------------------------
__global__ void gconv_agg_kernel(const float* __restrict__ H, float* __restrict__ AH,
                                 int Nj, int is_hand, int total)
{
  int idx = blockIdx.x * blockDim.x + threadIdx.x;
  if (idx >= total) return;
  const int* par = is_hand ? c_par_hand : c_par_body;
  int f = idx & 63;
  int i = (idx >> 6) % Nj;
  int g = idx / (64 * Nj);
  const float* hg = H + (size_t)g * Nj * 64;
  int pi = par[i];
  float acc = 0.f;
  for (int j = 0; j < Nj; ++j)
    if (pi == j || par[j] == i) acc += hg[(size_t)j * 64 + f];
  AH[idx] = acc;
}

// ---------------------------------------------------------------------------
// h = leaky_relu(LayerNorm(y)) + h ; one wave32 per 64-wide row, coalesced.
// ---------------------------------------------------------------------------
__global__ __launch_bounds__(256)
void ln_act_res_kernel(const float* __restrict__ Y, const float* __restrict__ g,
                       const float* __restrict__ b, float* __restrict__ H, int rows)
{
  int warp = (blockIdx.x * blockDim.x + threadIdx.x) >> 5;
  int lane = threadIdx.x & 31;
  if (warp >= rows) return;
  const float* y = Y + (size_t)warp * 64;
  float y0 = y[lane], y1 = y[lane + 32];
  float s = y0 + y1;
#pragma unroll
  for (int m = 16; m; m >>= 1) s += __shfl_xor(s, m, 32);
  float mean = s * (1.f / 64.f);
  float d0 = y0 - mean, d1 = y1 - mean;
  float vv = d0 * d0 + d1 * d1;
#pragma unroll
  for (int m = 16; m; m >>= 1) vv += __shfl_xor(vv, m, 32);
  float inv = rsqrtf(vv * (1.f / 64.f) + 1e-5f);
  float* h = H + (size_t)warp * 64;
  float z0 = d0 * inv * g[lane] + b[lane];
  float z1 = d1 * inv * g[lane + 32] + b[lane + 32];
  h[lane]      = leaky02(z0) + h[lane];
  h[lane + 32] = leaky02(z1) + h[lane + 32];
}

// ---------------------------------------------------------------------------
// Final LN over 256 + write into concat output [4096, 512] at coloff.
// One wave32 per row (8 elems/lane).
// ---------------------------------------------------------------------------
__global__ __launch_bounds__(256)
void final_ln_kernel(const float* __restrict__ P, const float* __restrict__ g,
                     const float* __restrict__ b, float* __restrict__ out, int coloff)
{
  int warp = (blockIdx.x * blockDim.x + threadIdx.x) >> 5;
  int lane = threadIdx.x & 31;
  if (warp >= G_TOT) return;
  const float* p = P + (size_t)warp * 256;
  float v[8];
  float s = 0.f;
#pragma unroll
  for (int i = 0; i < 8; ++i) { v[i] = p[lane + i * 32]; s += v[i]; }
#pragma unroll
  for (int m = 16; m; m >>= 1) s += __shfl_xor(s, m, 32);
  float mean = s * (1.f / 256.f);
  float vv = 0.f;
#pragma unroll
  for (int i = 0; i < 8; ++i) { float d = v[i] - mean; vv += d * d; }
#pragma unroll
  for (int m = 16; m; m >>= 1) vv += __shfl_xor(vv, m, 32);
  float inv = rsqrtf(vv * (1.f / 256.f) + 1e-5f);
  float* o = out + (size_t)warp * 512 + coloff;
#pragma unroll
  for (int i = 0; i < 8; ++i) {
    int c = lane + i * 32;
    o[c] = (v[i] - mean) * inv * g[c] + b[c];
  }
}

// ---------------------------------------------------------------------------
// Host side
// ---------------------------------------------------------------------------
static inline void launch_gemm(hipStream_t s, const float* A, const _Float16* Bp,
                               const float* bias, float* C, int M, int K, int N, int acc)
{
  dim3 grid(M / 16, N / 64);
  if (acc) gemm_wmma_kernel<1><<<grid, 128, 0, s>>>(A, Bp, bias, C, M, K, N);
  else     gemm_wmma_kernel<0><<<grid, 128, 0, s>>>(A, Bp, bias, C, M, K, N);
}

static inline void pack_w(hipStream_t s, const float* W, _Float16* Bp, int K, int N)
{
  int total = K * N;
  pack_w_kernel<<<(total + 255) / 256, 256, 0, s>>>(W, Bp, K, N, total);
}

struct StreamW {
  const float *bin, *asrc, *adst, *gatb, *gcb, *lng, *lnb, *bout, *ng, *nb;
  const _Float16 *Winp, *gatWp, *gcW1p, *gcW2p, *Woutp;   // packed f16 weights
};

static void run_stream(hipStream_t s, const float* X, const StreamW& w, int Nj,
                       int is_hand, int chunk, float* H, float* T1, float* T2,
                       float* P, float* out)
{
  // input projection: [4096,256] @ [256, Nj*64] + bin
  launch_gemm(s, X, w.Winp, w.bin, H, G_TOT, 256, Nj * JD, 0);

  const int rows = G_TOT * Nj;
  int gi = 0, ci = 0;
  const int kinds[5] = {0, 1, 0, 1, 0};  // 0=GAT, 1=GraphConv
  for (int li = 0; li < 5; ++li) {
    if (kinds[li] == 0) {
      for (int g0 = 0; g0 < G_TOT; g0 += chunk) {
        int CH = (G_TOT - g0 < chunk) ? (G_TOT - g0) : chunk;
        launch_gemm(s, H + (size_t)g0 * Nj * JD,
                    w.gatWp + (size_t)gi * JD * HEADS * JD,
                    nullptr, T2, CH * Nj, JD, HEADS * JD, 0);
        gat_attn_kernel<<<CH, 256, 0, s>>>(T2, w.asrc + gi * HEADS * JD,
                                           w.adst + gi * HEADS * JD,
                                           w.gatb + gi * JD, T1, Nj, g0, is_hand);
      }
      ++gi;
    } else {
      launch_gemm(s, H, w.gcW1p + (size_t)ci * JD * JD, w.gcb + ci * JD,
                  T1, rows, JD, JD, 0);
      int total = rows * JD;
      gconv_agg_kernel<<<(total + 255) / 256, 256, 0, s>>>(H, T2, Nj, is_hand, total);
      launch_gemm(s, T2, w.gcW2p + (size_t)ci * JD * JD, nullptr, T1, rows, JD, JD, 1);
      ++ci;
    }
    int th = rows * 32;
    ln_act_res_kernel<<<(th + 255) / 256, 256, 0, s>>>(
        T1, w.lng + li * JD, w.lnb + li * JD, H, rows);
  }

  // output projection + final LN into concat output
  launch_gemm(s, H, w.Woutp, w.bout, P, G_TOT, Nj * JD, 256, 0);
  int th = G_TOT * 32;
  final_ln_kernel<<<(th + 255) / 256, 256, 0, s>>>(P, w.ng, w.nb, out,
                                                   is_hand ? 256 : 0);
}

extern "C" void kernel_launch(void* const* d_in, const int* in_sizes, int n_in,
                              void* d_out, int out_size, void* d_ws, size_t ws_size,
                              hipStream_t stream)
{
  (void)in_sizes; (void)n_in; (void)out_size; (void)ws_size;

  const float* refined = (const float*)d_in[0];
  auto F = [&](int i) { return (const float*)d_in[i]; };

  // ----- workspace layout (floats) -----
  float* ws = (float*)d_ws;
  size_t off = 0;
  float* X  = ws + off; off += (size_t)G_TOT * 256;        //  4 MiB
  float* Hb = ws + off; off += (size_t)G_TOT * NB * JD;    // 10 MiB
  float* Hh = ws + off; off += (size_t)G_TOT * NH * JD;    // 42 MiB
  float* T1 = ws + off; off += (size_t)G_TOT * NH * JD;    // 42 MiB (Y buffer)
  float* T2 = ws + off; off += (size_t)G_TOT * NH * JD;    // 42 MiB (Hh-chunk / AH)
  float* P  = ws + off; off += (size_t)G_TOT * 256;        //  4 MiB
  _Float16* pk = (_Float16*)(ws + off);                    // packed weight arena
  size_t po = 0;
  auto alloc_pk = [&](size_t n) { _Float16* p = pk + po; po += n; return p; };

  // ----- pre-pack weights into WMMA fragment order -----
  struct Raw { int base; };
  const int bodyBase = 1, handBase = 16;
  StreamW sw[2];
  for (int s2 = 0; s2 < 2; ++s2) {
    int base = s2 ? handBase : bodyBase;
    int Nj   = s2 ? NH : NB;
    const float* Win  = F(base + 0);
    const float* gatW = F(base + 2);
    const float* gcW1 = F(base + 6);
    const float* gcW2 = F(base + 7);
    const float* Wout = F(base + 11);

    _Float16* Winp  = alloc_pk((size_t)256 * Nj * JD);
    _Float16* gatWp = alloc_pk((size_t)3 * JD * HEADS * JD);
    _Float16* gcW1p = alloc_pk((size_t)2 * JD * JD);
    _Float16* gcW2p = alloc_pk((size_t)2 * JD * JD);
    _Float16* Woutp = alloc_pk((size_t)Nj * JD * 256);

    pack_w(stream, Win, Winp, 256, Nj * JD);
    for (int gi = 0; gi < 3; ++gi)
      pack_w(stream, gatW + (size_t)gi * JD * HEADS * JD,
             gatWp + (size_t)gi * JD * HEADS * JD, JD, HEADS * JD);
    for (int ci = 0; ci < 2; ++ci) {
      pack_w(stream, gcW1 + (size_t)ci * JD * JD, gcW1p + (size_t)ci * JD * JD, JD, JD);
      pack_w(stream, gcW2 + (size_t)ci * JD * JD, gcW2p + (size_t)ci * JD * JD, JD, JD);
    }
    pack_w(stream, Wout, Woutp, Nj * JD, 256);

    sw[s2] = StreamW{F(base + 1), F(base + 3), F(base + 4), F(base + 5),
                     F(base + 8), F(base + 9), F(base + 10), F(base + 12),
                     F(base + 13), F(base + 14),
                     Winp, gatWp, gcW1p, gcW2p, Woutp};
  }

  // ----- x = transpose(refined_feats) -----
  int total = G_TOT * 256;
  transpose_btc_kernel<<<(total + 255) / 256, 256, 0, stream>>>(refined, X, total);

  // body: GAT feature GEMM fits T2 in one pass; hand: chunk at 1024 graphs
  run_stream(stream, X, sw[0], NB, /*is_hand=*/0, /*chunk=*/G_TOT,
             Hb, T1, T2, P, (float*)d_out);
  run_stream(stream, X, sw[1], NH, /*is_hand=*/1, /*chunk=*/1024,
             Hh, T1, T2, P, (float*)d_out);
}
